// HybridSupportFeatureRefiner_25769804426
// MI455X (gfx1250) — matrix operand order repaired and checked by reading, not compiled, not run on hardware
//
#include <hip/hip_runtime.h>
#include <hip/hip_bf16.h>

typedef _Float16 v16h __attribute__((ext_vector_type(16)));
typedef _Float16 v8h  __attribute__((ext_vector_type(8)));
typedef float    v8f  __attribute__((ext_vector_type(8)));

#define CC   256
#define HW   4096
#define KF   768      // fused channels excluding dist column
#define XS   776      // padded LDS row stride in f16 elems (768 + 8)
#define NPIX 64       // pixels (N) per block

// ---------------- k0: prep: f32->f16 weight conversion + init flags ----------------
__global__ void k0_prep(const float* __restrict__ wfus, _Float16* __restrict__ W16,
                        unsigned* __restrict__ distmax, int* __restrict__ hasfg) {
  int i = blockIdx.x * 256 + threadIdx.x;
  if (i < CC * KF) {
    int m = i / KF, k = i - m * KF;
    W16[i] = (_Float16)wfus[m * 769 + k];
  }
  if (i < 8) { distmax[i] = 0u; hasfg[i] = 0; }
}

// ---------------- k1: per-column vertical nearest-fg distance ----------------
// m_s[b,y,x] = support_mask[b, 4y, 4x]  (nearest resize 256->64)
__global__ void k1_vcol(const float* __restrict__ mask, int* __restrict__ vcol,
                        int* __restrict__ hasfg) {
  int b = blockIdx.x;
  int x = threadIdx.x;                 // 0..63 column
  const float* mb = mask + (size_t)b * 65536;
  int vdown[64];
  int d = 100000;
  bool any = false;
#pragma unroll
  for (int y = 0; y < 64; ++y) {
    bool fg = mb[y * 1024 + x * 4] > 0.5f;
    any |= fg;
    d = fg ? 0 : (d < 90000 ? d + 1 : d);
    vdown[y] = d;
  }
  d = 100000;
#pragma unroll
  for (int y = 63; y >= 0; --y) {
    bool fg = mb[y * 1024 + x * 4] > 0.5f;
    d = fg ? 0 : (d < 90000 ? d + 1 : d);
    int v = vdown[y] < d ? vdown[y] : d;
    vcol[b * HW + y * 64 + x] = v < 1000 ? v : 1000;
  }
  if (any) atomicOr(&hasfg[b], 1);
}

// ---------------- k2: exact EDT per pixel = min_x' (v[y][x']^2 + (x-x')^2) ----------------
__global__ void k2_dist(const int* __restrict__ vcol, const int* __restrict__ hasfg,
                        float* __restrict__ dist_raw, unsigned* __restrict__ distmax) {
  __shared__ int vl[256];
  int b = blockIdx.x >> 4;
  int pbase = (blockIdx.x & 15) * 256;     // 4 full rows of 64
  int t = threadIdx.x;
  vl[t] = vcol[b * HW + pbase + t];
  __syncthreads();
  int r = t >> 6;                          // local row 0..3
  int x = t & 63;
  int best = 0x3fffffff;
#pragma unroll
  for (int xp = 0; xp < 64; ++xp) {
    int vv = vl[r * 64 + xp];
    int dx = x - xp;
    int d2 = vv * vv + dx * dx;
    best = d2 < best ? d2 : best;
  }
  float raw = hasfg[b] ? sqrtf((float)best) : 1.0f;
  dist_raw[b * HW + pbase + t] = raw;
  atomicMax(&distmax[b], __float_as_uint(raw));   // raw >= 0 -> bit-pattern max OK
}

// ---------------- k3: fused attn + f_w + 1x1-conv GEMM via V_WMMA_F32_16X16X32_F16 --------
// Block: 512 threads = 16 waves. N-tile = 64 pixels, each wave owns 16 output channels
// and carries 4 accumulators (16x64 of C). One A fragment feeds 4 WMMAs per K-step.
__global__ __launch_bounds__(512) void k3_gemm(
    const float* __restrict__ f_s, const float* __restrict__ supp,
    const _Float16* __restrict__ W16, const float* __restrict__ wfus,
    const float* __restrict__ dist_raw, const unsigned* __restrict__ distmax,
    float* __restrict__ out) {
  __shared__ _Float16 X[NPIX * XS];   // fused input tile, f16: 64 * 776 * 2 = 99328 B
  __shared__ float part[8 * NPIX];
  __shared__ float attn[NPIX];
  __shared__ float dloc[NPIX];

  const int t   = threadIdx.x;
  const int b   = blockIdx.x >> 6;          // 64 tiles per sample
  const int p0  = (blockIdx.x & 63) * NPIX;
  const int pix = t & 63;
  const int cg  = t >> 6;                   // 0..7 channel group

  const float* fsb = f_s  + (size_t)b * CC * HW + p0 + pix;
  const float* spb = supp + (size_t)b * CC * HW + p0 + pix;

  // Stage f_s / supp tiles to LDS (f16) and accumulate attn dot product.
  float pa = 0.f;
#pragma unroll
  for (int it = 0; it < 32; ++it) {
    int c = cg + it * 8;
    float fv = fsb[(size_t)c * HW];
    float sv = spb[(size_t)c * HW];
    X[pix * XS + c]       = (_Float16)fv;
    X[pix * XS + 256 + c] = (_Float16)sv;
    pa += fv * sv;
  }
  part[cg * NPIX + pix] = pa;
  __syncthreads();
  if (t < NPIX) {
    float s = 0.f;
#pragma unroll
    for (int j = 0; j < 8; ++j) s += part[j * NPIX + t];
    attn[t] = 1.f / (1.f + __expf(-s));
    float mx = __uint_as_float(distmax[b]);
    dloc[t] = dist_raw[b * HW + p0 + t] / (mx + 1e-6f);
  }
  __syncthreads();
  {
    float av = attn[pix];
#pragma unroll
    for (int it = 0; it < 32; ++it) {
      int c = cg + it * 8;
      X[pix * XS + 512 + c] = (_Float16)((float)X[pix * XS + c] * av);
    }
  }
  __syncthreads();

  // WMMA: 4x (16x16) tiles per wave, K = 768 in 24 steps, shared A fragment.
  const int lane = t & 31;
  const int lo = lane & 15, hi = lane >> 4;
  const int m0 = (t >> 5) * 16;             // wave's output-channel base
  const _Float16* Wr = W16 + (size_t)(m0 + lo) * KF;  // A row M = lo
  const _Float16* Xr0 = &X[(0  + lo) * XS];           // B col N = j*16 + lo
  const _Float16* Xr1 = &X[(16 + lo) * XS];
  const _Float16* Xr2 = &X[(32 + lo) * XS];
  const _Float16* Xr3 = &X[(48 + lo) * XS];
  v8f acc0 = {0.f,0.f,0.f,0.f,0.f,0.f,0.f,0.f};
  v8f acc1 = acc0, acc2 = acc0, acc3 = acc0;
#pragma unroll 2
  for (int k0 = 0; k0 < KF; k0 += 32) {
    // A layout: elems 0..15 -> K = k0 + hi*16 + j
    v8h a0 = *(const v8h*)(Wr + k0 + hi * 16);
    v8h a1 = *(const v8h*)(Wr + k0 + hi * 16 + 8);
    v16h A = __builtin_shufflevector(a0, a1, 0,1,2,3,4,5,6,7,8,9,10,11,12,13,14,15);
    // B layout: elems 0..15 -> K = k0 + hi*16 + j, N = lo (per sub-tile)
    v8h b00 = *(const v8h*)(Xr0 + k0 + hi * 16);
    v8h b01 = *(const v8h*)(Xr0 + k0 + hi * 16 + 8);
    v8h b10 = *(const v8h*)(Xr1 + k0 + hi * 16);
    v8h b11 = *(const v8h*)(Xr1 + k0 + hi * 16 + 8);
    v8h b20 = *(const v8h*)(Xr2 + k0 + hi * 16);
    v8h b21 = *(const v8h*)(Xr2 + k0 + hi * 16 + 8);
    v8h b30 = *(const v8h*)(Xr3 + k0 + hi * 16);
    v8h b31 = *(const v8h*)(Xr3 + k0 + hi * 16 + 8);
    v16h B0 = __builtin_shufflevector(b00, b01, 0,1,2,3,4,5,6,7,8,9,10,11,12,13,14,15);
    v16h B1 = __builtin_shufflevector(b10, b11, 0,1,2,3,4,5,6,7,8,9,10,11,12,13,14,15);
    v16h B2 = __builtin_shufflevector(b20, b21, 0,1,2,3,4,5,6,7,8,9,10,11,12,13,14,15);
    v16h B3 = __builtin_shufflevector(b30, b31, 0,1,2,3,4,5,6,7,8,9,10,11,12,13,14,15);
    acc0 = __builtin_amdgcn_wmma_f32_16x16x32_f16(false, A, false, B0, (short)0, acc0, false, false);
    acc1 = __builtin_amdgcn_wmma_f32_16x16x32_f16(false, A, false, B1, (short)0, acc1, false, false);
    acc2 = __builtin_amdgcn_wmma_f32_16x16x32_f16(false, A, false, B2, (short)0, acc2, false, false);
    acc3 = __builtin_amdgcn_wmma_f32_16x16x32_f16(false, A, false, B3, (short)0, acc3, false, false);
  }

  // dist column (K = 768) in exact f32, then store y. C/D layout: M = v + hi*8, N = lo.
  float wd[8];
#pragma unroll
  for (int v = 0; v < 8; ++v) wd[v] = wfus[(m0 + v + hi * 8) * 769 + 768];
  float* ob = out + (size_t)b * (CC * HW) + (size_t)(m0 + hi * 8) * HW + p0 + lo;
  float dv0 = dloc[0 + lo], dv1 = dloc[16 + lo], dv2 = dloc[32 + lo], dv3 = dloc[48 + lo];
#pragma unroll
  for (int v = 0; v < 8; ++v) {
    ob[(size_t)v * HW + 0 ] = acc0[v] + wd[v] * dv0;
    ob[(size_t)v * HW + 16] = acc1[v] + wd[v] * dv1;
    ob[(size_t)v * HW + 32] = acc2[v] + wd[v] * dv2;
    ob[(size_t)v * HW + 48] = acc3[v] + wd[v] * dv3;
  }
}

// ---------------- k4: per-channel BN stats -> folded scale/shift (deterministic) ----------
__global__ void k4_stats(const float* __restrict__ y, const float* __restrict__ gamma,
                         const float* __restrict__ beta, float* __restrict__ scale,
                         float* __restrict__ shift) {
  __shared__ float rs[256], rs2[256];
  int m = blockIdx.x, t = threadIdx.x;
  float s = 0.f, s2 = 0.f;
  for (int i = t; i < 32768; i += 256) {
    int b = i >> 12, p = i & 4095;
    float v = y[(size_t)b * (CC * HW) + (size_t)m * HW + p];
    s += v; s2 += v * v;
  }
  rs[t] = s; rs2[t] = s2;
  __syncthreads();
  for (int off = 128; off > 0; off >>= 1) {
    if (t < off) { rs[t] += rs[t + off]; rs2[t] += rs2[t + off]; }
    __syncthreads();
  }
  if (t == 0) {
    float mean = rs[0] * (1.f / 32768.f);
    float var  = rs2[0] * (1.f / 32768.f) - mean * mean;   // biased, matches jnp.var
    float sc = gamma[m] * rsqrtf(var + 1e-5f);
    scale[m] = sc;
    shift[m] = beta[m] - mean * sc;
  }
}

// ---------------- k5: in-place BN apply + ReLU, float4 vectorized ----------------
__global__ void k5_bn(float* __restrict__ y, const float* __restrict__ scale,
                      const float* __restrict__ shift) {
  int i4 = blockIdx.x * 256 + threadIdx.x;   // 2,097,152 float4's
  int c = (i4 >> 10) & 255;
  float4 v = ((const float4*)y)[i4];
  float sc = scale[c], sh = shift[c];
  v.x = fmaxf(0.f, v.x * sc + sh);
  v.y = fmaxf(0.f, v.y * sc + sh);
  v.z = fmaxf(0.f, v.z * sc + sh);
  v.w = fmaxf(0.f, v.w * sc + sh);
  ((float4*)y)[i4] = v;
}

extern "C" void kernel_launch(void* const* d_in, const int* in_sizes, int n_in,
                              void* d_out, int out_size, void* d_ws, size_t ws_size,
                              hipStream_t stream) {
  (void)in_sizes; (void)n_in; (void)out_size; (void)ws_size;
  const float* f_s   = (const float*)d_in[0];
  const float* supp  = (const float*)d_in[1];
  const float* mask  = (const float*)d_in[2];
  const float* wfus  = (const float*)d_in[3];
  const float* gamma = (const float*)d_in[4];
  const float* beta  = (const float*)d_in[5];
  float* out = (float*)d_out;

  char* ws = (char*)d_ws;                         // layout (bytes):
  _Float16* W16      = (_Float16*)(ws);           //      0 : 256*768*2 = 393216
  float*    dist_raw = (float*)(ws + 393216);     // 393216 : 32768*4   = 131072
  int*      vcol     = (int*)  (ws + 524288);     // 524288 : 32768*4   = 131072
  unsigned* distmax  = (unsigned*)(ws + 655360);  // 655360 : 32
  int*      hasfg    = (int*)  (ws + 655392);     // 655392 : 32
  float*    scale    = (float*)(ws + 655424);     // 655424 : 1024
  float*    shift    = (float*)(ws + 656448);     // 656448 : 1024   (total ~657.5 KB)

  k0_prep<<<768, 256, 0, stream>>>(wfus, W16, distmax, hasfg);
  k1_vcol<<<8, 64, 0, stream>>>(mask, vcol, hasfg);
  k2_dist<<<128, 256, 0, stream>>>(vcol, hasfg, dist_raw, distmax);
  k3_gemm<<<512, 512, 0, stream>>>(f_s, supp, W16, wfus, dist_raw, distmax, out);
  k4_stats<<<256, 256, 0, stream>>>(out, gamma, beta, scale, shift);
  k5_bn<<<8192, 256, 0, stream>>>(out, scale, shift);
}
